// SocialLSTM_8907762171989
// MI455X (gfx1250) — matrix operand order repaired and checked by reading, not compiled, not run on hardware
//
#include <hip/hip_runtime.h>
#include <math.h>

// ---------------- constants ----------------
#define B        2
#define OBS_LEN  8
#define PRED_LEN 12
#define NPID     1024
#define D        128      // LSTM_DIM
#define E64      64       // EMB_DIM
#define G        64       // 8x8 grid
#define NROWS    (B * NPID)          // 2048
#define PN       (G * E64)           // 4096, proj GEMM N
#define LSTMK    256                 // [e|a|h]
#define LSTMN    512                 // 4*D

#define PROJ_LDB 136                 // 128 + 8 pad (halfs) -> stride = 68 dwords
#define LSTM_LDB 264                 // 256 + 8 pad (halfs) -> stride = 132 dwords

typedef __attribute__((ext_vector_type(16))) _Float16 v16h;
typedef __attribute__((ext_vector_type(8)))  _Float16 v8h;
typedef __attribute__((ext_vector_type(8)))  float    v8f;
typedef __attribute__((ext_vector_type(4)))  int      v4i;

// -------- CDNA5 async global->LDS staging (guarded; sync fallback) --------
#if defined(__has_builtin)
#  if __has_builtin(__builtin_amdgcn_global_load_async_to_lds_b128)
#    define HAVE_ASYNC_LDS 1
#  else
#    define HAVE_ASYNC_LDS 0
#  endif
#else
#  define HAVE_ASYNC_LDS 0
#endif

__device__ __forceinline__ void copy16_g2s(const _Float16* __restrict__ g, _Float16* s) {
#if HAVE_ASYNC_LDS
  // signature (from compiler diagnostic): param0 = v4i* in global AS, 4 args
  __builtin_amdgcn_global_load_async_to_lds_b128(
      (__attribute__((address_space(1))) v4i*)(v4i*)(const void*)g,
      (__attribute__((address_space(3))) v4i*)(v4i*)(void*)s, 0, 0);
#else
  *(v8h*)s = *(const v8h*)g;
#endif
}

__device__ __forceinline__ void wait_async_copies() {
#if HAVE_ASYNC_LDS
#  if __has_builtin(__builtin_amdgcn_s_wait_asynccnt)
  __builtin_amdgcn_s_wait_asynccnt(0);
#  else
  asm volatile("s_wait_asynccnt 0x0" ::: "memory");
#  endif
#endif
}

// ---------------- WMMA helpers ----------------
__device__ __forceinline__ v8f wmma32(v16h a, v16h b, v8f c) {
  // D = A(16x32 f16) * B(32x16 f16) + C(16x16 f32)
  return __builtin_amdgcn_wmma_f32_16x16x32_f16(false, a, false, b, (short)0, c,
                                                false, false);
}

// 32B fragment from two aligned 16B loads (works for global and LDS pointers)
__device__ __forceinline__ v16h load_frag16(const _Float16* p) {
  v8h lo = *(const v8h*)p;
  v8h hi = *(const v8h*)(p + 8);
  return __builtin_shufflevector(lo, hi, 0,1,2,3,4,5,6,7,8,9,10,11,12,13,14,15);
}

// A fragment: 16x32 f16 tile at Abase (row stride lda), contiguous k order, matching the B reads:
// lane L: row m=L&15; halves 0..15 = K (L>=16?16:0)+0..15
__device__ __forceinline__ v16h load_a_frag(const _Float16* Abase, int lda, int lane) {
  int m    = lane & 15;
  int koff = (lane >> 4) << 4;
  const _Float16* p = Abase + (size_t)m * lda + koff;
  v8h lo = *(const v8h*)(p);
  v8h hi = *(const v8h*)(p + 8);
  return __builtin_shufflevector(lo, hi, 0,1,2,3,4,5,6,7,8,9,10,11,12,13,14,15);
}

// ---------------- weight repack (f32 -> f16, N-major B layout) ----------------
// Bt_wa[n=g*64+e][k=d] = Wa[(g*128+d)*64 + e]    (4096 x 128)
__global__ void prep_wa(const float* __restrict__ Wa, _Float16* __restrict__ Bt) {
  int idx = blockIdx.x * 256 + threadIdx.x;           // n*128 + k
  int k = idx & 127, n = idx >> 7;
  int g = n >> 6, e = n & 63;
  Bt[idx] = (_Float16)Wa[(size_t)(g * 128 + k) * 64 + e];
}

// Bt_kr[n][k] : k<128 -> K[k][n] ; else R[k-128][n]   (512 x 256)
__global__ void prep_kr(const float* __restrict__ K, const float* __restrict__ R,
                        _Float16* __restrict__ Bt) {
  int idx = blockIdx.x * 256 + threadIdx.x;           // n*256 + k
  int k = idx & 255, n = idx >> 8;
  float v = (k < 128) ? K[(size_t)k * 512 + n] : R[(size_t)(k - 128) * 512 + n];
  Bt[idx] = (_Float16)v;
}

__global__ void init_hc(float* __restrict__ h, float* __restrict__ c) {
  size_t i = (size_t)blockIdx.x * 128 + threadIdx.x;
  h[i] = 0.f; c[i] = 0.f;
}

// ---------------- per-step: h->f16 into Acat cols 128..255, build x_t --------
__global__ void prestep(int t, const float* __restrict__ x, const float* __restrict__ noise,
                        const float* __restrict__ obuf, const float* __restrict__ h,
                        _Float16* __restrict__ Acat, float* __restrict__ xcur) {
  int row = blockIdx.x;        // 0..2047
  int d   = threadIdx.x;       // 0..127
  Acat[(size_t)row * LSTMK + 128 + d] = (_Float16)h[(size_t)row * D + d];
  int b = row >> 10, i = row & 1023;
  if (t < OBS_LEN) {
    if (d < 2)
      xcur[row * 2 + d] = x[(((size_t)b * OBS_LEN + t) * NPID + i) * 2 + d];
  } else if (d == 0) {
    const float* o = obuf + (size_t)row * 5;
    float sx = expf(o[2]), sy = expf(o[3]), rho = tanhf(o[4]);
    const float* zz = noise + (((size_t)b * PRED_LEN + (t - OBS_LEN)) * NPID + i) * 2;
    xcur[row * 2 + 0] = o[0] + sx * zz[0];
    xcur[row * 2 + 1] = o[1] + sy * (rho * zz[0] + sqrtf(fmaxf(0.f, 1.f - rho * rho)) * zz[1]);
  }
}

// ---------------- proj GEMM: P(1024x4096) = h_f16(1024x128) @ Bt_wa^T --------
// Block = 8 waves = 8 M-tiles (128 rows) x one 64-col N-group; B tile staged
// in LDS once per block via async-to-LDS. Grid: 8 mgroups x 64 ngroups = 512.
__global__ void __launch_bounds__(256) proj_gemm(const _Float16* __restrict__ A,
                                                 const _Float16* __restrict__ Bt,
                                                 float* __restrict__ P) {
  __shared__ __align__(32) _Float16 Bs[64 * PROJ_LDB];   // ~17 KB
  int tid = threadIdx.x;
  int ngroup = blockIdx.x & 63;   // 0..63
  int mgroup = blockIdx.x >> 6;   // 0..7
  int n0 = ngroup * 64;

  // stage B tile: 64 rows x 128 halfs = 16 KB, 16B chunks
#pragma unroll
  for (int it = 0; it < 4; ++it) {
    int chunk = it * 256 + tid;          // 0..1023
    int row = chunk >> 4;
    int co  = (chunk & 15) * 8;          // half offset, 16B granularity
    copy16_g2s(Bt + (size_t)(n0 + row) * 128 + co, &Bs[row * PROJ_LDB + co]);
  }

  // preload all A fragments (K = 128 -> 4 frags) while the copy is in flight
  int wid = tid >> 5, lane = tid & 31;
  int m0 = mgroup * 128 + wid * 16;
  const _Float16* Ab = A + (size_t)m0 * LSTMK;
  v16h afrag[4];
#pragma unroll
  for (int kc = 0; kc < 4; ++kc) afrag[kc] = load_a_frag(Ab + kc * 32, LSTMK, lane);

  wait_async_copies();
  __syncthreads();

  v8f acc[4] = {};
#pragma unroll
  for (int kc = 0; kc < 4; ++kc) {
    int kb = kc * 32 + ((lane >> 4) << 4);
#pragma unroll
    for (int nt = 0; nt < 4; ++nt) {
      v16h b = load_frag16(&Bs[(nt * 16 + (lane & 15)) * PROJ_LDB + kb]);
      acc[nt] = wmma32(afrag[kc], b, acc[nt]);
    }
  }

  int hi8 = (lane >> 4) << 3;
  int col = lane & 15;
#pragma unroll
  for (int v = 0; v < 8; ++v) {
    size_t r = (size_t)(m0 + hi8 + v) * PN + n0 + col;
    P[r +  0] = acc[0][v];
    P[r + 16] = acc[1][v];
    P[r + 32] = acc[2][v];
    P[r + 48] = acc[3][v];
  }
}

// ---------------- social gather: a_i = relu(sum_j P[j, cell(i,j)*64+e] + ba) --
// also computes e-embedding relu(x@We+be). 256 blocks x 256 thr (4 i's x 64 e).
__global__ void __launch_bounds__(256) social_gather(const float* __restrict__ xcur,
                                                     const float* __restrict__ P,
                                                     const float* __restrict__ We,
                                                     const float* __restrict__ be,
                                                     const float* __restrict__ ba,
                                                     _Float16* __restrict__ Acat, int b) {
  __shared__ float psx[256], psy[256];
  int tid = threadIdx.x;
  int wi = tid >> 6, e = tid & 63;
  int i   = blockIdx.x * 4 + wi;
  int row = b * NPID + i;
  float pix = xcur[row * 2 + 0], piy = xcur[row * 2 + 1];
  float acc = 0.f;
  for (int chunk = 0; chunk < NPID; chunk += 256) {
    __syncthreads();
    int jg = b * NPID + chunk + tid;
    psx[tid] = xcur[jg * 2 + 0];
    psy[tid] = xcur[jg * 2 + 1];
    __syncthreads();
#pragma unroll 4
    for (int jj = 0; jj < 256; ++jj) {
      int j = chunk + jj;
      float rx = psx[jj] - pix, ry = psy[jj] - piy;
      int cx = (int)floorf((rx + 2.0f) * 2.0f);
      int cy = (int)floorf((ry + 2.0f) * 2.0f);
      bool ok = (cx >= 0) & (cx < 8) & (cy >= 0) & (cy < 8) & (j != i);
      if (ok) {
        int idx = cx * 8 + cy;
        acc += P[(size_t)j * PN + idx * 64 + e];
      }
    }
  }
  float a = fmaxf(acc + ba[e], 0.f);
  float ee = fmaxf(pix * We[e] + piy * We[64 + e] + be[e], 0.f);
  Acat[(size_t)row * LSTMK + e]      = (_Float16)ee;
  Acat[(size_t)row * LSTMK + 64 + e] = (_Float16)a;
}

// ---------------- LSTM GEMM: Z(2048x512) = Acat(2048x256) @ Bt_kr^T ----------
// Block = 8 waves = 8 M-tiles x one 64-col N-group; B tile (64x256, 32 KB) in
// LDS. Grid: 16 mgroups x 8 ngroups = 128 blocks.
__global__ void __launch_bounds__(256) lstm_gemm(const _Float16* __restrict__ A,
                                                 const _Float16* __restrict__ Bt,
                                                 float* __restrict__ Z) {
  __shared__ __align__(32) _Float16 Bs[64 * LSTM_LDB];   // ~33 KB
  int tid = threadIdx.x;
  int ngroup = blockIdx.x & 7;    // 0..7
  int mgroup = blockIdx.x >> 3;   // 0..15
  int n0 = ngroup * 64;

  // stage B tile: 64 rows x 256 halfs = 32 KB
#pragma unroll
  for (int it = 0; it < 8; ++it) {
    int chunk = it * 256 + tid;          // 0..2047
    int row = chunk >> 5;
    int co  = (chunk & 31) * 8;
    copy16_g2s(Bt + (size_t)(n0 + row) * LSTMK + co, &Bs[row * LSTM_LDB + co]);
  }

  int wid = tid >> 5, lane = tid & 31;
  int m0 = mgroup * 128 + wid * 16;
  const _Float16* Ab = A + (size_t)m0 * LSTMK;
  v16h afrag[8];
#pragma unroll
  for (int kc = 0; kc < 8; ++kc) afrag[kc] = load_a_frag(Ab + kc * 32, LSTMK, lane);

  wait_async_copies();
  __syncthreads();

  v8f acc[4] = {};
#pragma unroll
  for (int kc = 0; kc < 8; ++kc) {
    int kb = kc * 32 + ((lane >> 4) << 4);
#pragma unroll
    for (int nt = 0; nt < 4; ++nt) {
      v16h b = load_frag16(&Bs[(nt * 16 + (lane & 15)) * LSTM_LDB + kb]);
      acc[nt] = wmma32(afrag[kc], b, acc[nt]);
    }
  }

  int hi8 = (lane >> 4) << 3;
  int col = lane & 15;
#pragma unroll
  for (int v = 0; v < 8; ++v) {
    size_t r = (size_t)(m0 + hi8 + v) * LSTMN + n0 + col;
    Z[r +  0] = acc[0][v];
    Z[r + 16] = acc[1][v];
    Z[r + 32] = acc[2][v];
    Z[r + 48] = acc[3][v];
  }
}

// ---------------- gates + output head ----------------
__device__ __forceinline__ float sigmf(float x) { return 1.0f / (1.0f + expf(-x)); }

__global__ void __launch_bounds__(256) gate(int t, const float* __restrict__ Z,
                                            const float* __restrict__ bb,
                                            float* __restrict__ c, float* __restrict__ h,
                                            const float* __restrict__ Wp,
                                            const float* __restrict__ bp,
                                            float* __restrict__ obuf,
                                            float* __restrict__ out) {
  __shared__ float hs[2][D];
  int li = threadIdx.x >> 7, d = threadIdx.x & 127;
  int row = blockIdx.x * 2 + li;
  const float* z = Z + (size_t)row * LSTMN;
  float zi = z[d]       + bb[d];
  float zf = z[128 + d] + bb[128 + d];
  float zg = z[256 + d] + bb[256 + d];
  float zo = z[384 + d] + bb[384 + d];
  float ig = sigmf(zi), fg = sigmf(zf), gg = tanhf(zg), og = sigmf(zo);
  size_t hd = (size_t)row * D + d;
  float cn = fg * c[hd] + ig * gg;
  float hn = og * tanhf(cn);
  c[hd] = cn; h[hd] = hn;
  hs[li][d] = hn;
  __syncthreads();
  if (d < 5) {
    float s = bp[d];
#pragma unroll 8
    for (int k = 0; k < D; ++k) s += hs[li][k] * Wp[k * 5 + d];
    obuf[(size_t)row * 5 + d] = s;
    if (t >= OBS_LEN) {
      int b = row >> 10, i = row & 1023;
      out[(((size_t)b * PRED_LEN + (t - OBS_LEN)) * NPID + i) * 5 + d] = s;
    }
  }
}

// ---------------- host orchestration ----------------
extern "C" void kernel_launch(void* const* d_in, const int* in_sizes, int n_in,
                              void* d_out, int out_size, void* d_ws, size_t ws_size,
                              hipStream_t stream) {
  const float* x     = (const float*)d_in[0];
  const float* We    = (const float*)d_in[1];
  const float* be    = (const float*)d_in[2];
  const float* Wa    = (const float*)d_in[3];
  const float* ba    = (const float*)d_in[4];
  const float* Kw    = (const float*)d_in[5];
  const float* Rw    = (const float*)d_in[6];
  const float* bb    = (const float*)d_in[7];
  const float* Wp    = (const float*)d_in[8];
  const float* bp    = (const float*)d_in[9];
  const float* noise = (const float*)d_in[10];
  float* out = (float*)d_out;

  char* ws = (char*)d_ws;
  _Float16* BT_WA = (_Float16*)(ws + 0x0000000);                 // 4096*128*2 = 1 MB
  _Float16* BT_KR = (_Float16*)(ws + 0x0100000);                 // 512*256*2  = 256 KB
  _Float16* ACAT  = (_Float16*)(ws + 0x0140000);                 // 2048*256*2 = 1 MB
  float*    P     = (float*)   (ws + 0x0240000);                 // 1024*4096*4 = 16 MB
  float*    Zb    = (float*)   (ws + 0x1240000);                 // 2048*512*4 = 4 MB
  float*    H     = (float*)   (ws + 0x1640000);                 // 1 MB
  float*    C     = (float*)   (ws + 0x1740000);                 // 1 MB
  float*    XCUR  = (float*)   (ws + 0x1840000);                 // 16 KB
  float*    OBUF  = (float*)   (ws + 0x1844000);                 // 40 KB

  prep_wa<<<(PN * 128) / 256, 256, 0, stream>>>(Wa, BT_WA);
  prep_kr<<<(LSTMN * LSTMK) / 256, 256, 0, stream>>>(Kw, Rw, BT_KR);
  init_hc<<<NROWS, 128, 0, stream>>>(H, C);

  for (int t = 0; t < OBS_LEN + PRED_LEN; ++t) {
    prestep<<<NROWS, 128, 0, stream>>>(t, x, noise, OBUF, H, ACAT, XCUR);
    for (int b = 0; b < B; ++b) {
      const _Float16* Ah = ACAT + (size_t)b * NPID * LSTMK + 128;  // h cols
      proj_gemm<<<512, 256, 0, stream>>>(Ah, BT_WA, P);
      social_gather<<<NPID / 4, 256, 0, stream>>>(XCUR, P, We, be, ba, ACAT, b);
    }
    lstm_gemm<<<128, 256, 0, stream>>>(ACAT, BT_KR, Zb);
    gate<<<NROWS / 2, 256, 0, stream>>>(t, Zb, bb, C, H, Wp, bp, OBUF, out);
  }
}
